// KAN_20615843021188
// MI455X (gfx1250) — matrix operand order repaired and checked by reading, not compiled, not run on hardware
//
#include <hip/hip_runtime.h>

// ---------------------------------------------------------------------------
// KAN forward for MI455X (gfx1250, wave32, WMMA).
// Reformulated as GEMM: H_out = (DM [B x 1280]) @ (W [1280 x 256]) / 256
// where DM[b, c*256+i] = JacobiP_c(tanh(LN(h)[b,i])), W[c*256+i, o] = coefs[i,o,c].
// f16 operands + f32 accumulation via v_wmma_f32_16x16x32_f16.
// ---------------------------------------------------------------------------

typedef __attribute__((ext_vector_type(16))) _Float16 v16h;
typedef __attribute__((ext_vector_type(8)))  float    v8f;
typedef __attribute__((ext_vector_type(4)))  float    v4f;

#define IN_DIM   256
#define N_COEF   5
#define KDIM     (IN_DIM * N_COEF)   // 1280
#define KTILES   (KDIM / 32)         // 40
#define NTILES   (IN_DIM / 16)       // 16
#define BROWS    16                  // rows (M) per block
#define LN_EPS   1e-6f

// ---------------------------------------------------------------------------
// Repack coefs[layer][i][o][c] (f32) into B-fragment register layout (f16):
// unit u = ((layer*KTILES + kt)*NTILES + nt)*32 + lane, 16 halfs per unit.
// Within a unit, element e (= 2*j + h) holds W[k, n] with
//   k = kt*32 + ((lane>=16)?16:0) + e,   n = nt*16 + (lane&15)
// contiguous k order for the B (32x16) operand, the same order kan_layer's A fragments in LDS use.
// ---------------------------------------------------------------------------
__global__ void kan_repack(const float* __restrict__ coefs,
                           _Float16* __restrict__ bpack) {
    int u = blockIdx.x * blockDim.x + threadIdx.x;
    const int total = 2 * KTILES * NTILES * 32;
    if (u >= total) return;
    int layer = u / (KTILES * NTILES * 32);
    int r     = u % (KTILES * NTILES * 32);
    int kt    = r / (NTILES * 32);
    int r2    = r % (NTILES * 32);
    int nt    = r2 / 32;
    int lane  = r2 % 32;
    int n     = nt * 16 + (lane & 15);
    int kbase = kt * 32 + ((lane >> 4) << 4);
    const float* cl = coefs + (size_t)layer * IN_DIM * IN_DIM * N_COEF;

    _Float16 tmp[16];
#pragma unroll
    for (int e = 0; e < 16; ++e) {
        int k = kbase + e;
        int c = k >> 8;      // k / 256
        int i = k & 255;     // k % 256
        tmp[e] = (_Float16)cl[(size_t)i * (IN_DIM * N_COEF) + n * N_COEF + c];
    }
    *((v16h*)bpack + u) = *(v16h*)tmp;
}

// ---------------------------------------------------------------------------
// Fused layer: LayerNorm + tanh + Jacobi features -> LDS (A-fragment layout),
// then WMMA GEMM against pre-packed B fragments.
// Block: 256 threads (8 waves), 16 rows of the batch, full N=256.
// ---------------------------------------------------------------------------
__global__ __launch_bounds__(256)
void kan_layer(const float* __restrict__ hin,
               const _Float16* __restrict__ bpack,   // this layer's pack
               const float* __restrict__ alphas, int layer,
               const float* __restrict__ ln_scale,   // already + layer*256
               const float* __restrict__ ln_bias,    // already + layer*256
               float* __restrict__ hout) {
    // A fragments for 16 rows x 1280 K, f16, pre-swizzled: 40 KB
    __shared__ __align__(32) _Float16 smem[KTILES * 32 * 16];

    const int tid  = threadIdx.x;
    const int m    = tid >> 4;    // 0..15 : row within tile
    const int part = tid & 15;    // 16 lanes per row, 16 dims each
    const int grow = blockIdx.x * BROWS + m;

    // ---------------- Phase 1: LN + tanh + Jacobi -> LDS ----------------
    float v[16];
    {
        const float* src = hin + (size_t)grow * IN_DIM + part * 16;
#pragma unroll
        for (int q = 0; q < 4; ++q) {
            v4f t = *(const v4f*)(src + q * 4);
#pragma unroll
            for (int e = 0; e < 4; ++e) v[q * 4 + e] = t[e];
        }
    }
    float s = 0.f;
#pragma unroll
    for (int e = 0; e < 16; ++e) s += v[e];
#pragma unroll
    for (int off = 1; off < 16; off <<= 1) s += __shfl_xor(s, off, 32);
    const float mu = s * (1.0f / 256.0f);

    float sq = 0.f;
#pragma unroll
    for (int e = 0; e < 16; ++e) { float d = v[e] - mu; sq += d * d; }
#pragma unroll
    for (int off = 1; off < 16; off <<= 1) sq += __shfl_xor(sq, off, 32);
    const float rstd = rsqrtf(sq * (1.0f / 256.0f) + LN_EPS);

    // Jacobi recurrence coefficients (depend only on a = tanh(alpha))
    const float a = tanhf(alphas[layer]);
    float B2, C2, iA2, B3, C3, iA3, B4, C4, iA4;
    { float t = 4.f + 2.f * a;
      iA2 = 1.f / (4.f * (2.f + 2.f * a) * (t - 2.f));
      B2  = (t - 1.f) * t * (t - 2.f);
      C2  = 2.f * (1.f + a) * (1.f + a) * t; }
    { float t = 6.f + 2.f * a;
      iA3 = 1.f / (6.f * (3.f + 2.f * a) * (t - 2.f));
      B3  = (t - 1.f) * t * (t - 2.f);
      C3  = 2.f * (2.f + a) * (2.f + a) * t; }
    { float t = 8.f + 2.f * a;
      iA4 = 1.f / (8.f * (4.f + 2.f * a) * (t - 2.f));
      B4  = (t - 1.f) * t * (t - 2.f);
      C4  = 2.f * (3.f + a) * (3.f + a) * t; }

#pragma unroll
    for (int e = 0; e < 16; ++e) {
        const int d = part * 16 + e;
        float hn = (v[e] - mu) * rstd * ln_scale[d] + ln_bias[d];
        float x  = tanhf(hn);
        float P0 = 1.f;
        float P1 = (a + 1.f) * x;
        float P2 = (B2 * x * P1 - C2 * P0) * iA2;
        float P3 = (B3 * x * P2 - C3 * P1) * iA3;
        float P4 = (B4 * x * P3 - C4 * P2) * iA4;
        // A-fragment (16-bit, 16x32) lane/VGPR mapping; since 256 % 32 == 0,
        // kk = (c*256+d) & 31 = d & 31 is identical for all 5 coefficients.
        const int kk   = d & 31;
        const int grp  = kk >> 3;
        const int lane = m + ((kk >> 4) << 4);
        const int jj   = (kk & 15) >> 1;
        const int h    = kk & 1;
        // half index = (kt*32 + lane)*16 + jj*2 + h, kt = c*8 + d/32
        const int base = ((d >> 5) * 32 + lane) * 16 + jj * 2 + h;
        smem[base + 0 * 4096] = (_Float16)P0;   // c stride: 8 ktiles * 512 halfs
        smem[base + 1 * 4096] = (_Float16)P1;
        smem[base + 2 * 4096] = (_Float16)P2;
        smem[base + 3 * 4096] = (_Float16)P3;
        smem[base + 4 * 4096] = (_Float16)P4;
    }

    __syncthreads();

    // ---------------- Phase 2: WMMA GEMM ----------------
    const int w    = tid >> 5;       // wave 0..7
    const int lane = tid & 31;
    const int nt0  = w * 2;          // each wave: 2 adjacent N tiles

    v8f acc0 = {0.f, 0.f, 0.f, 0.f, 0.f, 0.f, 0.f, 0.f};
    v8f acc1 = {0.f, 0.f, 0.f, 0.f, 0.f, 0.f, 0.f, 0.f};
    const v16h* bp = (const v16h*)bpack;

#pragma unroll 4
    for (int kt = 0; kt < KTILES; ++kt) {
        if (kt + 1 < KTILES) {
            __builtin_prefetch(&bp[((kt + 1) * NTILES + nt0) * 32 + lane], 0, 1);
        }
        v16h af = *(const v16h*)(smem + (kt * 32 + lane) * 16);
        v16h b0 = bp[(kt * NTILES + nt0    ) * 32 + lane];
        v16h b1 = bp[(kt * NTILES + nt0 + 1) * 32 + lane];
        acc0 = __builtin_amdgcn_wmma_f32_16x16x32_f16(false, af, false, b0,
                                                      (short)0, acc0, false, false);
        acc1 = __builtin_amdgcn_wmma_f32_16x16x32_f16(false, af, false, b1,
                                                      (short)0, acc1, false, false);
    }

    // Epilogue: C/D 16x16 f32 layout: VGPR vv, lanes0-15 -> M=vv, lanes16-31 -> M=vv+8
    const int rbase = blockIdx.x * BROWS + ((lane >> 4) << 3);
    const int col0  = nt0 * 16 + (lane & 15);
#pragma unroll
    for (int vv = 0; vv < 8; ++vv) {
        const size_t rowoff = (size_t)(rbase + vv) * IN_DIM;
        hout[rowoff + col0]      = acc0[vv] * (1.0f / 256.0f);
        hout[rowoff + col0 + 16] = acc1[vv] * (1.0f / 256.0f);
    }
}

// ---------------------------------------------------------------------------
extern "C" void kernel_launch(void* const* d_in, const int* in_sizes, int n_in,
                              void* d_out, int out_size, void* d_ws, size_t ws_size,
                              hipStream_t stream) {
    const float* x        = (const float*)d_in[0];
    const float* coefs    = (const float*)d_in[1];
    const float* alphas   = (const float*)d_in[2];
    const float* ln_scale = (const float*)d_in[3];
    const float* ln_bias  = (const float*)d_in[4];
    float* out = (float*)d_out;

    // Workspace layout: [bpack: 2*1280*256 f16 = 1.25 MB][h1: 8192*256 f32 = 8 MB]
    _Float16* bpack = (_Float16*)d_ws;
    float* h1 = (float*)((char*)d_ws + (size_t)2 * KDIM * IN_DIM * sizeof(_Float16));

    const int repack_units = 2 * KTILES * NTILES * 32;   // 40960
    kan_repack<<<(repack_units + 255) / 256, 256, 0, stream>>>(coefs, bpack);

    const int batch   = in_sizes[0] / IN_DIM;            // 8192
    const int nblocks = batch / BROWS;                   // 512

    kan_layer<<<nblocks, 256, 0, stream>>>(x, bpack, alphas, 0,
                                           ln_scale, ln_bias, h1);
    kan_layer<<<nblocks, 256, 0, stream>>>(h1, bpack + (size_t)KDIM * IN_DIM, alphas, 1,
                                           ln_scale + IN_DIM, ln_bias + IN_DIM, out);
}
